// spectrograms_24584392802508
// MI455X (gfx1250) — matrix operand order repaired and checked by reading, not compiled, not run on hardware
//
#include <hip/hip_runtime.h>
#include <hip/hip_bf16.h>
#include <math.h>

// ---------------- types ----------------
typedef __bf16 v2bf  __attribute__((ext_vector_type(2)));
typedef __bf16 v8bf  __attribute__((ext_vector_type(8)));
typedef __bf16 v16bf __attribute__((ext_vector_type(16)));
typedef float  v8f   __attribute__((ext_vector_type(8)));
typedef unsigned long long u64;

#define WMMA_BF16(a,b,c) __builtin_amdgcn_wmma_f32_16x16x32_bf16(false,(a),false,(b),(short)0,(c),false,false)

// ---------------- problem constants ----------------
constexpr int C_WIN   = 16384;
constexpr int C_D     = 4096;
constexpr int C_HOP   = 512;
constexpr int C_NREG  = 25;
constexpr int C_TWOK  = 1024;
constexpr int C_B     = 128;
constexpr int C_MROWS = C_B * C_NREG;     // 3200
constexpr int C_KHALF = 512;
constexpr int C_K2    = 128;
constexpr int C_NR2Y  = 193;
constexpr int C_K3    = 256;
constexpr int C_CONV2K= 3200;             // 128*25
constexpr int C_HCOLS = C_K3 * C_NR2Y;    // 49408
constexpr int H1_STRIDE = 3208;           // conflict-free, 16B aligned
constexpr int AS_STR  = 88;               // A tile LDS row stride (44 DW: conflict-free)
constexpr int BS_STR  = 80;               // B tile LDS row stride

// ---------------- CDNA5 async copy helpers ----------------
__device__ __forceinline__ unsigned lds_off(const void* p) { return (unsigned)(size_t)p; }
__device__ __forceinline__ void async_b128(u64 gaddr, unsigned lds) {
  asm volatile("global_load_async_to_lds_b128 %0, %1, off"
               :: "v"(lds), "v"(gaddr) : "memory");
}
__device__ __forceinline__ void wait_async0() {
  asm volatile("s_wait_asynccnt 0x0" ::: "memory");
}

// ---------------- fragment helpers ----------------
__device__ __forceinline__ v16bf frag_cat(v8bf lo, v8bf hi) {
  return __builtin_shufflevector(lo, hi, 0,1,2,3,4,5,6,7,8,9,10,11,12,13,14,15);
}
__device__ __forceinline__ v16bf load_fragA(const __bf16* row, int kbase) {
  v8bf lo = *(const v8bf*)(row + kbase);
  v8bf hi = *(const v8bf*)(row + kbase + 8);
  return frag_cat(lo, hi);
}

// ---------------- prep kernels ----------------
__global__ __launch_bounds__(256) void prep_cvt_kernel(const float* __restrict__ in,
                                                       __bf16* __restrict__ out, int n) {
  int i = blockIdx.x * 256 + threadIdx.x;
  if (i < n) out[i] = (__bf16)in[i];
}

__global__ __launch_bounds__(256) void prep_filtersT_kernel(const float* __restrict__ in,
                                                            __bf16* __restrict__ out) {
  int i = blockIdx.x * 256 + threadIdx.x;
  int d = i & (C_D - 1);
  int n = i >> 12;
  out[(size_t)n * C_D + d] = (__bf16)in[(size_t)d * C_TWOK + n];
}

__global__ __launch_bounds__(256) void prep_w2t_kernel(const float* __restrict__ in,
                                                       __bf16* __restrict__ out) {
  int i = blockIdx.x * 256 + threadIdx.x;
  int k3  = i / C_CONV2K;
  int rem = i - k3 * C_CONV2K;
  int dx  = rem >> 7;
  int c   = rem & 127;
  out[i] = (__bf16)in[(size_t)k3 * C_CONV2K + c * 25 + dx];
}

// ---------------- stage 1: spec = windows @ filters ----------------
// M=3200 (b*25+r), N=1024, K=4096. Double-buffered, K-step 64.
// A: gathered f32 from x -> cvt -> ds_store. B: async global->LDS.
__global__ __launch_bounds__(256) void gemm_spec_kernel(const float* __restrict__ x,
                                                        const __bf16* __restrict__ Bt,
                                                        float* __restrict__ C) {
  __shared__ __bf16 As[2][64 * AS_STR];
  __shared__ __bf16 Bs[2][128 * BS_STR];
  const int tid  = threadIdx.x;
  const int lane = tid & 31, wave = tid >> 5;
  const int hi = lane >> 4, lm = lane & 15;
  const int m0 = blockIdx.x * 64, n0 = blockIdx.y * 128;
  const int wm = (wave >> 2) * 32, wn = (wave & 3) * 32;

  const int arow = tid >> 2, akc = (tid & 3) * 16;  // A: 64 rows x 4 chunks of 16
  const int brow = tid >> 1, bkc = (tid & 1) * 32;  // B: 128 rows x 2 chunks of 32
  const int aoff = arow * AS_STR + akc;             // LDS element offset (keeps DS addrspace)
  const int mg = m0 + arow;
  const int bb = mg / C_NREG;
  const int rr = mg - bb * C_NREG;
  const float*  ag = x  + (size_t)bb * C_WIN + rr * C_HOP + akc;
  const __bf16* bg = Bt + (size_t)(n0 + brow) * C_D + bkc;
  const unsigned bl[2] = { lds_off(&Bs[0][brow * BS_STR + bkc]),
                           lds_off(&Bs[1][brow * BS_STR + bkc]) };

  auto fill_B = [&](int buf, int k0) {
    u64 g = (u64)(bg + k0);
    unsigned l = bl[buf];
    async_b128(g, l); async_b128(g + 16, l + 16);
    async_b128(g + 32, l + 32); async_b128(g + 48, l + 48);
  };
  auto fill_A = [&](int buf, int k0) {
    float4 f0 = *(const float4*)(ag + k0);
    float4 f1 = *(const float4*)(ag + k0 + 4);
    float4 f2 = *(const float4*)(ag + k0 + 8);
    float4 f3 = *(const float4*)(ag + k0 + 12);
    v8bf t0, t1;
    t0[0]=(__bf16)f0.x; t0[1]=(__bf16)f0.y; t0[2]=(__bf16)f0.z; t0[3]=(__bf16)f0.w;
    t0[4]=(__bf16)f1.x; t0[5]=(__bf16)f1.y; t0[6]=(__bf16)f1.z; t0[7]=(__bf16)f1.w;
    t1[0]=(__bf16)f2.x; t1[1]=(__bf16)f2.y; t1[2]=(__bf16)f2.z; t1[3]=(__bf16)f2.w;
    t1[4]=(__bf16)f3.x; t1[5]=(__bf16)f3.y; t1[6]=(__bf16)f3.z; t1[7]=(__bf16)f3.w;
    *(v8bf*)(&As[buf][aoff])     = t0;   // stays in DS pipe (ds_store_b128)
    *(v8bf*)(&As[buf][aoff + 8]) = t1;
  };

  v8f acc[2][2] = {};
  fill_B(0, 0);
  fill_A(0, 0);
  const int nIter = C_D / 64;
  for (int it = 0; it < nIter; ++it) {
    const int cur = it & 1;
    wait_async0();
    __syncthreads();
    if (it + 1 < nIter) { fill_B(cur ^ 1, (it + 1) * 64); fill_A(cur ^ 1, (it + 1) * 64); }
    #pragma unroll
    for (int ks = 0; ks < 2; ++ks) {
      v16bf a0 = load_fragA(&As[cur][(wm      + lm) * AS_STR], ks * 32 + hi * 16);
      v16bf a1 = load_fragA(&As[cur][(wm + 16 + lm) * AS_STR], ks * 32 + hi * 16);
      v16bf b0 = *(const v16bf*)(&Bs[cur][(wn      + lm) * BS_STR + ks * 32 + hi * 16]);
      v16bf b1 = *(const v16bf*)(&Bs[cur][(wn + 16 + lm) * BS_STR + ks * 32 + hi * 16]);
      acc[0][0] = WMMA_BF16(a0, b0, acc[0][0]);
      acc[0][1] = WMMA_BF16(a0, b1, acc[0][1]);
      acc[1][0] = WMMA_BF16(a1, b0, acc[1][0]);
      acc[1][1] = WMMA_BF16(a1, b1, acc[1][1]);
    }
    __syncthreads();
  }
  #pragma unroll
  for (int mi = 0; mi < 2; ++mi)
    #pragma unroll
    for (int ni = 0; ni < 2; ++ni)
      #pragma unroll
      for (int v = 0; v < 8; ++v) {
        int m = m0 + wm + mi * 16 + v + 8 * hi;
        int n = n0 + wn + ni * 16 + lm;
        C[(size_t)m * C_TWOK + n] = acc[mi][ni][v];
      }
}

// ---------------- stage 2: zx = log(cos^2 + sin^2 + 1e-14) -> bf16 ----------------
__global__ __launch_bounds__(256) void epilogue_zx_kernel(const float* __restrict__ spec,
                                                          __bf16* __restrict__ zx) {
  int i = blockIdx.x * 256 + threadIdx.x;
  int row = i >> 9;
  int c   = i & 511;
  float2 cs = *(const float2*)(spec + (size_t)row * C_TWOK + 2 * c);
  float v = logf(cs.x * cs.x + cs.y * cs.y + 1e-14f);
  zx[i] = (__bf16)v;
}

// ---------------- stage 3: fused conv1(+relu) -> LDS -> conv2(+relu) ----------------
__global__ __launch_bounds__(256) void fused_conv_kernel(const __bf16* __restrict__ zx,
                                                         const __bf16* __restrict__ w1,
                                                         const __bf16* __restrict__ w2t,
                                                         __bf16* __restrict__ h2) {
  extern __shared__ __align__(16) char smem_raw[];
  __bf16* zxs = (__bf16*)smem_raw;                 // 25*512
  __bf16* h1s = zxs + C_NREG * C_KHALF;            // 16 * H1_STRIDE

  const int tid  = threadIdx.x;
  const int lane = tid & 31, wave = tid >> 5;
  const int hi = lane >> 4, lm = lane & 15;
  const int b  = blockIdx.y;
  const int jb = blockIdx.x * 16;

  // async-stage the full 25x512 zx slab for this batch (1600 x 16B chunks)
  const u64 zg = (u64)(zx + (size_t)b * (C_NREG * C_KHALF));
  const unsigned zl = lds_off(zxs);
  for (int c = tid; c < (C_NREG * C_KHALF) / 8; c += 256)
    async_b128(zg + (u64)c * 16, zl + (unsigned)c * 16);
  wait_async0();
  __syncthreads();

  // ---- conv1: wave handles channel tile c = wave*16 + lm ----
  const int c1 = wave * 16 + lm;
  for (int r = 0; r < C_NREG; ++r) {
    const __bf16* zrow = zxs + r * C_KHALF;
    v8f acc = {};
    for (int ks = 0; ks < C_K2; ks += 32) {
      int cbase = 2 * jb + 2 * lm + ks + hi * 16;   // sliding stride-2 window
      v16bf a;
      #pragma unroll
      for (int q = 0; q < 4; ++q) {
        v2bf t0 = *(const v2bf*)(zrow + cbase + 2 * q);
        v2bf t1 = *(const v2bf*)(zrow + cbase + 8 + 2 * q);
        a[2*q]   = t0[0]; a[2*q+1]   = t0[1];
        a[8+2*q] = t1[0]; a[8+2*q+1] = t1[1];
      }
      v16bf bfrag = *(const v16bf*)(w1 + c1 * C_K2 + ks + hi * 16);
      acc = WMMA_BF16(a, bfrag, acc);
    }
    #pragma unroll
    for (int v = 0; v < 8; ++v) {
      float f = acc[v];
      f = f > 0.0f ? f : 0.0f;
      h1s[(v + 8 * hi) * H1_STRIDE + r * C_K2 + c1] = (__bf16)f;  // [j][r*128+c]
    }
  }
  __syncthreads();

  // ---- conv2: rows j (16), K = 3200, N-tiles (wave*32 + {0,16}) ----
  const int n0 = wave * 32;
  v8f acc0 = {}, acc1 = {};
  const __bf16* arowbase = h1s + lm * H1_STRIDE;
  const __bf16* b0base   = w2t + (size_t)(n0 + lm) * C_CONV2K;
  for (int ks = 0; ks < C_CONV2K; ks += 32) {
    v16bf a  = load_fragA(arowbase + ks, hi * 16);
    v16bf b0 = *(const v16bf*)(b0base + ks + hi * 16);
    acc0 = WMMA_BF16(a, b0, acc0);
    v16bf b1 = *(const v16bf*)(b0base + (size_t)16 * C_CONV2K + ks + hi * 16);
    acc1 = WMMA_BF16(a, b1, acc1);
  }
  #pragma unroll
  for (int t = 0; t < 2; ++t) {
    const v8f& acc = t ? acc1 : acc0;
    int k3 = n0 + t * 16 + lm;
    #pragma unroll
    for (int v = 0; v < 8; ++v) {
      int j = jb + v + 8 * hi;
      if (j < C_NR2Y) {
        float f = acc[v];
        f = f > 0.0f ? f : 0.0f;
        h2[(size_t)b * C_HCOLS + k3 * C_NR2Y + j] = (__bf16)f;
      }
    }
  }
}

// ---------------- generic bf16 GEMM (fully async-to-LDS, double-buffered) ----------------
__global__ __launch_bounds__(256) void gemm_bf16_kernel(const __bf16* __restrict__ A,
                                                        const __bf16* __restrict__ Bt,
                                                        float* __restrict__ C,
                                                        int Kdim, int lda, int ldb, int ldc) {
  __shared__ __bf16 As[2][64 * AS_STR];
  __shared__ __bf16 Bs[2][128 * BS_STR];
  const int tid  = threadIdx.x;
  const int lane = tid & 31, wave = tid >> 5;
  const int hi = lane >> 4, lm = lane & 15;
  const int m0 = blockIdx.x * 64, n0 = blockIdx.y * 128;
  const int wm = (wave >> 2) * 32, wn = (wave & 3) * 32;

  const int arow = tid >> 2, akc = (tid & 3) * 16;
  const int brow = tid >> 1, bkc = (tid & 1) * 32;
  const __bf16* ag = A  + (size_t)(m0 + arow) * lda + akc;
  const __bf16* bg = Bt + (size_t)(n0 + brow) * ldb + bkc;
  const unsigned al[2] = { lds_off(&As[0][arow * AS_STR + akc]),
                           lds_off(&As[1][arow * AS_STR + akc]) };
  const unsigned bl[2] = { lds_off(&Bs[0][brow * BS_STR + bkc]),
                           lds_off(&Bs[1][brow * BS_STR + bkc]) };

  auto fill = [&](int buf, int k0) {
    u64 ga = (u64)(ag + k0);
    async_b128(ga, al[buf]); async_b128(ga + 16, al[buf] + 16);
    u64 gb = (u64)(bg + k0);
    async_b128(gb, bl[buf]);           async_b128(gb + 16, bl[buf] + 16);
    async_b128(gb + 32, bl[buf] + 32); async_b128(gb + 48, bl[buf] + 48);
  };

  v8f acc[2][2] = {};
  fill(0, 0);
  const int nIter = Kdim / 64;
  for (int it = 0; it < nIter; ++it) {
    const int cur = it & 1;
    wait_async0();
    __syncthreads();
    if (it + 1 < nIter) fill(cur ^ 1, (it + 1) * 64);
    #pragma unroll
    for (int ks = 0; ks < 2; ++ks) {
      v16bf a0 = load_fragA(&As[cur][(wm      + lm) * AS_STR], ks * 32 + hi * 16);
      v16bf a1 = load_fragA(&As[cur][(wm + 16 + lm) * AS_STR], ks * 32 + hi * 16);
      v16bf b0 = *(const v16bf*)(&Bs[cur][(wn      + lm) * BS_STR + ks * 32 + hi * 16]);
      v16bf b1 = *(const v16bf*)(&Bs[cur][(wn + 16 + lm) * BS_STR + ks * 32 + hi * 16]);
      acc[0][0] = WMMA_BF16(a0, b0, acc[0][0]);
      acc[0][1] = WMMA_BF16(a0, b1, acc[0][1]);
      acc[1][0] = WMMA_BF16(a1, b0, acc[1][0]);
      acc[1][1] = WMMA_BF16(a1, b1, acc[1][1]);
    }
    __syncthreads();
  }
  #pragma unroll
  for (int mi = 0; mi < 2; ++mi)
    #pragma unroll
    for (int ni = 0; ni < 2; ++ni)
      #pragma unroll
      for (int v = 0; v < 8; ++v) {
        int m = m0 + wm + mi * 16 + v + 8 * hi;
        int n = n0 + wn + ni * 16 + lm;
        C[(size_t)m * ldc + n] = acc[mi][ni][v];
      }
}

// ---------------- host launch ----------------
extern "C" void kernel_launch(void* const* d_in, const int* in_sizes, int n_in,
                              void* d_out, int out_size, void* d_ws, size_t ws_size,
                              hipStream_t stream) {
  (void)in_sizes; (void)n_in; (void)out_size; (void)ws_size;
  const float* x       = (const float*)d_in[0];
  const float* filters = (const float*)d_in[1];
  const float* conv1_w = (const float*)d_in[2];
  const float* conv2_w = (const float*)d_in[3];
  const float* beta_w  = (const float*)d_in[4];
  float* out = (float*)d_out;

  char* ws = (char*)d_ws;
  size_t off = 0;
  auto alloc = [&](size_t bytes) -> void* {
    void* p = ws + off;
    off = (off + bytes + 255) & ~(size_t)255;
    return p;
  };
  __bf16* filtersT = (__bf16*)alloc((size_t)C_TWOK * C_D * 2);
  float*  spec     = (float*) alloc((size_t)C_MROWS * C_TWOK * 4);
  __bf16* zx       = (__bf16*)alloc((size_t)C_MROWS * C_KHALF * 2);
  __bf16* w1       = (__bf16*)alloc((size_t)C_K2 * C_K2 * 2);
  __bf16* w2t      = (__bf16*)alloc((size_t)C_K3 * C_CONV2K * 2);
  __bf16* betab    = (__bf16*)alloc((size_t)C_B * C_HCOLS * 2);
  __bf16* h2       = (__bf16*)alloc((size_t)C_B * C_HCOLS * 2);

  prep_filtersT_kernel<<<(C_TWOK * C_D) / 256, 256, 0, stream>>>(filters, filtersT);
  prep_cvt_kernel<<<(C_K2 * C_K2) / 256, 256, 0, stream>>>(conv1_w, w1, C_K2 * C_K2);
  prep_w2t_kernel<<<(C_K3 * C_CONV2K) / 256, 256, 0, stream>>>(conv2_w, w2t);
  prep_cvt_kernel<<<(C_B * C_HCOLS) / 256, 256, 0, stream>>>(beta_w, betab, C_B * C_HCOLS);

  dim3 g1(C_MROWS / 64, C_TWOK / 128);       // (50, 8)
  gemm_spec_kernel<<<g1, 256, 0, stream>>>(x, filtersT, spec);

  epilogue_zx_kernel<<<(C_MROWS * C_KHALF) / 256, 256, 0, stream>>>(spec, zx);

  dim3 g2(13, C_B);
  size_t smem = (size_t)C_NREG * C_KHALF * 2 + (size_t)16 * H1_STRIDE * 2;  // ~128 KB
  fused_conv_kernel<<<g2, 256, smem, stream>>>(zx, w1, w2t, h2);

  dim3 g3(C_B / 64, 1);                      // M=128, N=128, K=49408
  gemm_bf16_kernel<<<g3, 256, 0, stream>>>(h2, betab, out, C_HCOLS, C_HCOLS, C_HCOLS, 128);
}
